// GaussianDiffusion_Beta_63196148793838
// MI455X (gfx1250) — matrix operand order repaired and checked by reading, not compiled, not run on hardware
//
#include <hip/hip_runtime.h>
#include <math.h>

#define TSTEPS 20

typedef __attribute__((ext_vector_type(16))) _Float16 v16h;
typedef __attribute__((ext_vector_type(8)))  _Float16 v8h;
typedef __attribute__((ext_vector_type(8)))  float    v8f;

__global__ __launch_bounds__(128)
void diffusion_sample_kernel(const float* __restrict__ x_init,
                             const float* __restrict__ noises,
                             const float* __restrict__ W1, const float* __restrict__ b1,
                             const float* __restrict__ W2, const float* __restrict__ b2,
                             const float* __restrict__ W3, const float* __restrict__ b3,
                             float* __restrict__ out, int Btot)
{
    // W2 in f16 B-fragment order: [ (n*4+kb) ][ lane ][ 16 halves ]
    __shared__ __align__(32) _Float16 w2frag[32 * 32 * 16];        // 32 KB
    __shared__ __align__(16) float    w1t[128][8];                 // W1 cols + b1, padded
    __shared__ float b2s[128];
    __shared__ float s_sra[TSTEPS], s_srm1[TSTEPS], s_c1[TSTEPS], s_c2[TSTEPS], s_sig[TSTEPS];
    __shared__ __align__(16) _Float16 h2lds[4][16 * 136];          // padded stride vs banks
    __shared__ float xl[4][16][4];                                 // per-wave x tiles

    const int tid   = threadIdx.x;
    const int wave  = tid >> 5;
    const int lane  = tid & 31;
    const int row   = lane & 15;               // C/D column, A row
    const int base0 = (lane < 16) ? 0 : 8;     // A-fragment K sub-offset
    const int kboff = (lane < 16) ? 0 : 16;    // unused; B-fragments also use base0
    const int g0    = (blockIdx.x * 4 + wave) * 16;

    // ---- beta schedule (matches numpy float64 path) ----
    if (tid == 0) {
        const double s  = 0.008;
        const double PI = 3.14159265358979323846;
        double ac[TSTEPS + 1];
        for (int i = 0; i <= TSTEPS; ++i) {
            double v = cos(((((double)i) / TSTEPS) + s) / (1.0 + s) * PI * 0.5);
            ac[i] = v * v;
        }
        double cum = 1.0;
        for (int t = 0; t < TSTEPS; ++t) {
            double beta = 1.0 - ac[t + 1] / ac[t];
            beta = beta < 0.0 ? 0.0 : (beta > 0.999 ? 0.999 : beta);
            double alpha = 1.0 - beta;
            double acp = cum;           // cumprod before step t
            cum *= alpha;
            s_sra[t]  = (float)sqrt(1.0 / cum);
            s_srm1[t] = (float)sqrt(1.0 / cum - 1.0);
            s_c1[t]   = (float)(beta * sqrt(acp) / (1.0 - cum));
            s_c2[t]   = (float)((1.0 - acp) * sqrt(alpha) / (1.0 - cum));
            double var = beta * (1.0 - acp) / (1.0 - cum);
            s_sig[t]  = (float)sqrt(var < 1e-20 ? 1e-20 : var);
        }
    }

    // ---- stage W1^T (+b1), b2 ----
    for (int j = tid; j < 128; j += blockDim.x) {
        for (int r5 = 0; r5 < 5; ++r5) w1t[j][r5] = W1[r5 * 128 + j];
        w1t[j][5] = b1[j];
        w1t[j][6] = 0.f; w1t[j][7] = 0.f;
        b2s[j] = b2[j];
    }

    // ---- W2 -> f16 B-fragments in LDS ----
    for (int idx = wave * 8; idx < wave * 8 + 8; ++idx) {
        const int n = idx >> 2, kb = idx & 3;
        #pragma unroll
        for (int h = 0; h < 16; ++h) {
            const int k = kb * 32 + ((h < 8) ? (base0 + h) : (base0 + 16 + (h - 8)));
            w2frag[(idx * 32 + lane) * 16 + h] = (_Float16)W2[k * 128 + n * 16 + row];
        }
    }

    // ---- W3 -> register B-fragments (cols >= 3 zero-padded), b3 ----
    v16h  b3f[4];
    const float bias3 = (row < 3) ? b3[row] : 0.f;
    #pragma unroll
    for (int kb = 0; kb < 4; ++kb) {
        #pragma unroll
        for (int h = 0; h < 16; ++h) {
            const int k = kb * 32 + ((h < 8) ? (base0 + h) : (base0 + 16 + (h - 8)));
            b3f[kb][h] = (row < 3) ? (_Float16)W3[k * 3 + row] : (_Float16)0.f;
        }
    }

    // ---- load x_init tile ----
    if (lane < 16) {
        const int g = g0 + lane;
        xl[wave][lane][0] = x_init[g * 3 + 0];
        xl[wave][lane][1] = x_init[g * 3 + 1];
        xl[wave][lane][2] = x_init[g * 3 + 2];
        xl[wave][lane][3] = 0.f;
    }
    __syncthreads();

    // ================= 20 reverse-diffusion steps =================
    for (int t = TSTEPS - 1; t >= 0; --t) {
        const float sra  = s_sra[t],  srm1 = s_srm1[t];
        const float c1   = s_c1[t],   c2   = s_c2[t],  sig = s_sig[t];
        const float tf   = (float)t / (float)TSTEPS;

        // ---- layer 1 (K=5) in VALU, directly into A-fragment layout ----
        const float x0 = xl[wave][row][0];
        const float x1 = xl[wave][row][1];
        const float x2 = xl[wave][row][2];
        v16h a[4];
        #pragma unroll
        for (int kb = 0; kb < 4; ++kb) {
            #pragma unroll
            for (int h = 0; h < 16; ++h) {
                const int k = kb * 32 + ((h < 8) ? (base0 + h) : (base0 + 16 + (h - 8)));
                const float4 w  = *(const float4*)&w1t[k][0];
                const float2 wb = *(const float2*)&w1t[k][4];
                float v = fmaf(x0, w.x, fmaf(x1, w.y, fmaf(x2, w.z,
                          fmaf(tf, w.w, wb.x + wb.y))));           // beta_cond = 1.0
                a[kb][h] = (_Float16)fmaxf(v, 0.f);
            }
        }

        // ---- layer 2: 8 N-tiles x 4 K-blocks of v_wmma_f32_16x16x32_f16 ----
        #pragma unroll
        for (int n = 0; n < 8; ++n) {
            const float bias = b2s[n * 16 + row];
            v8f acc = {bias, bias, bias, bias, bias, bias, bias, bias};
            #pragma unroll
            for (int kb = 0; kb < 4; ++kb) {
                const v16h bf = *(const v16h*)&w2frag[((n * 4 + kb) * 32 + lane) * 16];
                acc = __builtin_amdgcn_wmma_f32_16x16x32_f16(
                          false, a[kb], false, bf, (short)0, acc, false, false);
            }
            #pragma unroll
            for (int r = 0; r < 8; ++r) {   // D layout -> row-major f16 LDS (relu)
                h2lds[wave][(r + base0) * 136 + n * 16 + row] =
                    (_Float16)fmaxf(acc[r], 0.f);
            }
        }
        __syncthreads();   // h2 visible across lanes before transpose-read

        // ---- layer 3: read h2 back in A-fragment layout, 4 WMMAs ----
        v8f acc3 = {bias3, bias3, bias3, bias3, bias3, bias3, bias3, bias3};
        #pragma unroll
        for (int kb = 0; kb < 4; ++kb) {
            const v8h lo = *(const v8h*)&h2lds[wave][row * 136 + kb * 32 + base0];
            const v8h hi = *(const v8h*)&h2lds[wave][row * 136 + kb * 32 + 16 + base0];
            v16h a2;
            #pragma unroll
            for (int i = 0; i < 8; ++i) { a2[i] = lo[i]; a2[i + 8] = hi[i]; }
            acc3 = __builtin_amdgcn_wmma_f32_16x16x32_f16(
                       false, a2, false, b3f[kb], (short)0, acc3, false, false);
        }

        // ---- posterior update (lanes holding valid eps columns N<3) ----
        if (row < 3) {
            const int d = row;
            #pragma unroll
            for (int r = 0; r < 8; ++r) {
                const int m = r + base0;
                const int g = g0 + m;
                const float xo = xl[wave][m][d];
                const float e  = acc3[r];
                float xr = sra * xo - srm1 * e;
                xr = fminf(fmaxf(xr, -1.f), 1.f);
                if (d == 2) xr = fmaxf(xr, 0.f);   // energy col -> [0,1]
                float nz = 0.f;
                if (t > 0) nz = noises[((size_t)t * (size_t)Btot + (size_t)g) * 3 + d];
                xl[wave][m][d] = c1 * xr + c2 * xo + sig * nz;
            }
        }
        __syncthreads();   // x tile visible before next step's layer 1
    }

    // ---- write final x ----
    if (lane < 16) {
        const int g = g0 + lane;
        out[g * 3 + 0] = xl[wave][lane][0];
        out[g * 3 + 1] = xl[wave][lane][1];
        out[g * 3 + 2] = xl[wave][lane][2];
    }
}

extern "C" void kernel_launch(void* const* d_in, const int* in_sizes, int n_in,
                              void* d_out, int out_size, void* d_ws, size_t ws_size,
                              hipStream_t stream) {
    const float* x_init = (const float*)d_in[0];
    const float* noises = (const float*)d_in[1];
    const float* W1     = (const float*)d_in[2];
    const float* b1     = (const float*)d_in[3];
    const float* W2     = (const float*)d_in[4];
    const float* b2     = (const float*)d_in[5];
    const float* W3     = (const float*)d_in[6];
    const float* b3     = (const float*)d_in[7];
    float* out = (float*)d_out;

    const int Btot = in_sizes[0] / 3;        // 524288
    const int nblk = Btot / 64;              // 64 samples per block (4 waves x 16)

    diffusion_sample_kernel<<<dim3(nblk), dim3(128), 0, stream>>>(
        x_init, noises, W1, b1, W2, b2, W3, b3, out, Btot);
}
